// Attention_58944131170853
// MI455X (gfx1250) — matrix operand order repaired and checked by reading, not compiled, not run on hardware
//
#include <hip/hip_runtime.h>

typedef __bf16 bf16_t;
typedef bf16_t bf16x8 __attribute__((ext_vector_type(8)));
typedef bf16_t v16bf  __attribute__((ext_vector_type(16)));
typedef float  v8f    __attribute__((ext_vector_type(8)));
typedef int    v4i    __attribute__((ext_vector_type(4)));

#define GLOBAL_AS __attribute__((address_space(1)))
#define SHARED_AS __attribute__((address_space(3)))

#define DIM    320
#define HEADS  5
#define HD     64
#define NB     8       // batch
#define NTOK   4096    // 64*64 queries
#define MTOK   1024    // 32*32 keys
#define KCONV  1280    // 2*2*320

#if defined(__has_builtin)
#if __has_builtin(__builtin_amdgcn_global_load_async_to_lds_b128) && \
    __has_builtin(__builtin_amdgcn_s_wait_asynccnt)
#define HAVE_ASYNC_LDS 1
#endif
#endif
#ifndef HAVE_ASYNC_LDS
#define HAVE_ASYNC_LDS 0
#endif

__device__ __forceinline__ bf16_t f2bf(float f) { return (bf16_t)f; }

__device__ __forceinline__ v16bf mk_frag(bf16x8 lo, bf16x8 hi) {
  return __builtin_shufflevector(lo, hi, 0,1,2,3,4,5,6,7,8,9,10,11,12,13,14,15);
}

// A-fragment (16x32 bf16, row-major): lane row = lane%16, chunks at
// k + (lane/16)*8 and +16 (16-bit A layout, ISA 7.12.2).
__device__ __forceinline__ v16bf load_afrag(const bf16_t* p) {
  bf16x8 lo = *reinterpret_cast<const bf16x8*>(p);
  bf16x8 hi = *reinterpret_cast<const bf16x8*>(p + 16);
  return mk_frag(lo, hi);
}

// B-fragment (32x16 bf16) from Bt[N][K]: lane col = lane%16,
// chunks at k + (lane/16)*8 and +16 (same K order as A).
__device__ __forceinline__ v16bf load_bfrag(const bf16_t* p) {
  bf16x8 lo = *reinterpret_cast<const bf16x8*>(p);
  bf16x8 hi = *reinterpret_cast<const bf16x8*>(p + 16);
  return mk_frag(lo, hi);
}

__device__ __forceinline__ v8f wmma_bf16(v16bf a, v16bf b, v8f c) {
  return __builtin_amdgcn_wmma_f32_16x16x32_bf16(false, a, false, b, (short)0, c, false, false);
}

// 16-byte global -> LDS stage: async path if available, sync fallback.
__device__ __forceinline__ void stage16(const bf16_t* g, bf16_t* l) {
#if HAVE_ASYNC_LDS
  __builtin_amdgcn_global_load_async_to_lds_b128(
      (GLOBAL_AS v4i*)(g),
      (SHARED_AS v4i*)(l), 0, 0);
#else
  *reinterpret_cast<bf16x8*>(l) = *reinterpret_cast<const bf16x8*>(g);
#endif
}
__device__ __forceinline__ void stage_wait() {
#if HAVE_ASYNC_LDS
  __builtin_amdgcn_s_wait_asynccnt(0);
#endif
}

// ---------------------------------------------------------------- elementwise
__global__ __launch_bounds__(256) void cvt_f32_bf16(const float* __restrict__ s,
                                                    bf16_t* __restrict__ d, int n) {
  int i = blockIdx.x * 256 + threadIdx.x;
  if (i < n) d[i] = f2bf(s[i]);
}

// Wt[n*K + k] = bf16(W[k*N + n])
__global__ __launch_bounds__(256) void transpose_cvt(const float* __restrict__ W,
                                                     bf16_t* __restrict__ Wt,
                                                     int K, int N) {
  int idx = blockIdx.x * 256 + threadIdx.x;
  if (idx >= K * N) return;
  int k = idx / N, n = idx % N;
  Wt[(size_t)n * K + k] = f2bf(W[idx]);
}

// im2col: xp[(b*1024 + oy*32 + ox)*1280 + (dy*2+dx)*320 + c] = xbf[b,2oy+dy,2ox+dx,c]
__global__ __launch_bounds__(256) void patch_gather(const bf16_t* __restrict__ xbf,
                                                    bf16_t* __restrict__ xp) {
  int idx = blockIdx.x * 256 + threadIdx.x;
  if (idx >= NB * MTOK * KCONV) return;
  int kk = idx % KCONV;
  int r  = idx / KCONV;
  int b  = r >> 10;
  int oy = (r >> 5) & 31;
  int ox = r & 31;
  int p   = kk / DIM;
  int cin = kk % DIM;
  int dy = p >> 1, dx = p & 1;
  xp[idx] = xbf[(((size_t)b * 64 + (2 * oy + dy)) * 64 + (2 * ox + dx)) * DIM + cin];
}

// ---------------------------------------------------------------- generic GEMM
// out[M,N] = A[M,K](bf16 rowmaj) x Bt[N,K](bf16 rowmaj = B^T).
// Block: 256 threads = 8 waves; each wave owns a 32x64 tile -> block = 256x64.
// B panel (64 cols x 32 K) staged per K-step into double-buffered LDS with
// async-to-LDS issued one step ahead.
// mode 0: outf[r*N+n] = acc + bias[n]
// mode 1: outb[r*N+n] = bf16(acc)
// mode 2: kv split: cols 0..319 -> k[b,h,m,d]; 320..639 -> vT[b,h,d,m]
__global__ __launch_bounds__(256) void gemm_wmma(const bf16_t* __restrict__ A, int lda,
                                                 const bf16_t* __restrict__ Bt, int K,
                                                 int M, int N,
                                                 const float* __restrict__ bias,
                                                 float* __restrict__ outf,
                                                 bf16_t* __restrict__ outb,
                                                 bf16_t* __restrict__ kdst,
                                                 bf16_t* __restrict__ vTdst,
                                                 int mode) {
  __shared__ __align__(16) bf16_t bsh[2][64 * 32];

  const int t = threadIdx.x;
  const int lane = t & 31;
  const int wave = t >> 5;
  const int row0 = blockIdx.y * 256 + wave * 32;
  const int col0 = blockIdx.x * 64;
  const int hg = lane >> 4;
  const int c  = lane & 15;

  // staging map: thread t moves 8 bf16 (16B): row t/4 of the 64-col panel,
  // K-chunk (t%4)*8.
  const int brow = t >> 2;
  const int bcol = (t & 3) * 8;
  const bf16_t* bstage = Bt + (size_t)(col0 + brow) * K + bcol;
  bf16_t* lst[2] = { &bsh[0][brow * 32 + bcol], &bsh[1][brow * 32 + bcol] };

  const v8f vzero = {0.f,0.f,0.f,0.f,0.f,0.f,0.f,0.f};
  v8f acc[2][4];
#pragma unroll
  for (int rt = 0; rt < 2; ++rt)
#pragma unroll
    for (int nt = 0; nt < 4; ++nt) acc[rt][nt] = vzero;

  const bf16_t* arow0 = A + (size_t)(row0 + c) * lda + hg * 8;
  const bf16_t* arow1 = arow0 + (size_t)16 * lda;

  stage16(bstage, lst[0]);
  stage_wait();
  __syncthreads();

  for (int k0 = 0; k0 < K; k0 += 32) {
    const int buf = (k0 >> 5) & 1;
    const bf16_t* lbuf = &bsh[buf][0];
    if (k0 + 32 < K) stage16(bstage + k0 + 32, lst[buf ^ 1]);   // prefetch next panel

    v16bf a0 = load_afrag(arow0 + k0);
    v16bf a1 = load_afrag(arow1 + k0);
#pragma unroll
    for (int nt = 0; nt < 4; ++nt) {
      v16bf bfrag = load_bfrag(lbuf + (nt * 16 + c) * 32 + hg * 8);
      acc[0][nt] = wmma_bf16(a0, bfrag, acc[0][nt]);
      acc[1][nt] = wmma_bf16(a1, bfrag, acc[1][nt]);
    }
    stage_wait();
    __syncthreads();
  }

#pragma unroll
  for (int rt = 0; rt < 2; ++rt) {
#pragma unroll
    for (int nt = 0; nt < 4; ++nt) {
#pragma unroll
      for (int i = 0; i < 8; ++i) {
        int r = row0 + rt * 16 + i + hg * 8;   // D layout: VGPR i -> row i / i+8
        int n = col0 + nt * 16 + c;
        float v = acc[rt][nt][i];
        if (mode == 0) {
          outf[(size_t)r * N + n] = v + (bias ? bias[n] : 0.f);
        } else if (mode == 1) {
          outb[(size_t)r * N + n] = f2bf(v);
        } else {
          int b = r >> 10, mrow = r & 1023;    // M = NB*1024
          int pair = n / DIM;
          int nn = n % DIM;
          int h = nn / HD, d = nn % HD;
          if (pair == 0)
            kdst[(((size_t)(b * HEADS + h)) * MTOK + mrow) * HD + d] = f2bf(v);
          else
            vTdst[(((size_t)(b * HEADS + h)) * HD + d) * MTOK + mrow] = f2bf(v);
        }
      }
    }
  }
}

// ---------------------------------------------------------------- layernorm
// One wave per row of 320; 4 rows per 128-thread block.
__global__ __launch_bounds__(128) void layernorm_bf16(const float* __restrict__ y,
                                                      const float* __restrict__ g,
                                                      const float* __restrict__ bb,
                                                      bf16_t* __restrict__ out) {
  int row  = blockIdx.x * 4 + (threadIdx.x >> 5);
  int lane = threadIdx.x & 31;
  const float* yr = y + (size_t)row * DIM;
  float vals[10];
  float s = 0.f;
#pragma unroll
  for (int t = 0; t < 10; ++t) { vals[t] = yr[lane + 32 * t]; s += vals[t]; }
#pragma unroll
  for (int m = 1; m < 32; m <<= 1) s += __shfl_xor(s, m, 32);
  float mu = s * (1.0f / DIM);
  float vs = 0.f;
#pragma unroll
  for (int t = 0; t < 10; ++t) { float d = vals[t] - mu; vs += d * d; }
#pragma unroll
  for (int m = 1; m < 32; m <<= 1) vs += __shfl_xor(vs, m, 32);
  float rinv = rsqrtf(vs * (1.0f / DIM) + 1e-3f);
  bf16_t* orow = out + (size_t)row * DIM;
#pragma unroll
  for (int t = 0; t < 10; ++t) {
    int cidx = lane + 32 * t;
    orow[cidx] = f2bf((vals[t] - mu) * rinv * g[cidx] + bb[cidx]);
  }
}

// ---------------------------------------------------------------- attention
// One wave per (b, head, 32-query tile). Flash-style online softmax over
// M=1024 keys in 16 chunks of 64. blockDim = 32 (single wave).
__global__ __launch_bounds__(32) void attn_wmma(const bf16_t* __restrict__ qbuf,
                                                const bf16_t* __restrict__ kbuf,
                                                const bf16_t* __restrict__ vT,
                                                bf16_t* __restrict__ attnout) {
  __shared__ __align__(16) bf16_t pshm[32 * 64];

  const int lane = threadIdx.x;
  const int qt = blockIdx.x, h = blockIdx.y, b = blockIdx.z;
  const int hg = lane >> 4;
  const int c  = lane & 15;
  const float scale = 0.125f;   // hd^-0.5 = 1/8

  // q A-fragments: 2 row tiles x 2 K windows over hd=64
  const bf16_t* qbase = qbuf + ((size_t)(b * NTOK + qt * 32 + c)) * DIM + h * HD + hg * 8;
  v16bf qa[2][2];
#pragma unroll
  for (int rt = 0; rt < 2; ++rt)
#pragma unroll
    for (int kk = 0; kk < 2; ++kk)
      qa[rt][kk] = load_afrag(qbase + (size_t)rt * 16 * DIM + kk * 32);

  const bf16_t* kb = kbuf + ((size_t)(b * HEADS + h)) * MTOK * HD;
  const bf16_t* vb = vT   + ((size_t)(b * HEADS + h)) * HD * MTOK;

  const v8f vzero = {0.f,0.f,0.f,0.f,0.f,0.f,0.f,0.f};
  v8f O[2][4];
  float m_i[2][8], l_i[2][8];
#pragma unroll
  for (int rt = 0; rt < 2; ++rt) {
#pragma unroll
    for (int dt = 0; dt < 4; ++dt) O[rt][dt] = vzero;
#pragma unroll
    for (int i = 0; i < 8; ++i) { m_i[rt][i] = -1e30f; l_i[rt][i] = 0.f; }
  }

  for (int j = 0; j < MTOK / 64; ++j) {
    const int m0 = j * 64;
    if (j + 1 < MTOK / 64) {        // prefetch next key/value chunks (8KB each)
      __builtin_prefetch(kb + (size_t)(m0 + 64) * HD + lane * 128, 0, 3);
      __builtin_prefetch(vb + (size_t)lane * MTOK + m0 + 64, 0, 3);
    }

    // ---- S = scale * q @ k^T  (32 x 64), k-frags prefetched one tile ahead
    v8f S[2][4];
#pragma unroll
    for (int rt = 0; rt < 2; ++rt)
#pragma unroll
      for (int nt = 0; nt < 4; ++nt) S[rt][nt] = vzero;

    v16bf kf0[2], kf1[2];
    kf0[0] = load_bfrag(kb + (size_t)(m0 + c) * HD + hg * 8);
    kf0[1] = load_bfrag(kb + (size_t)(m0 + c) * HD + 32 + hg * 8);
#pragma unroll
    for (int nt = 0; nt < 4; ++nt) {
      if (nt < 3) {
        const bf16_t* kp = kb + (size_t)(m0 + (nt + 1) * 16 + c) * HD + hg * 8;
        kf1[0] = load_bfrag(kp);
        kf1[1] = load_bfrag(kp + 32);
      }
#pragma unroll
      for (int rt = 0; rt < 2; ++rt) {
        S[rt][nt] = wmma_bf16(qa[rt][0], kf0[0], S[rt][nt]);
        S[rt][nt] = wmma_bf16(qa[rt][1], kf0[1], S[rt][nt]);
      }
      kf0[0] = kf1[0]; kf0[1] = kf1[1];
    }

    // ---- online softmax stats per row tile
    float alpha[2][8];
#pragma unroll
    for (int rt = 0; rt < 2; ++rt) {
#pragma unroll
      for (int nt = 0; nt < 4; ++nt)
#pragma unroll
        for (int i = 0; i < 8; ++i) S[rt][nt][i] *= scale;
#pragma unroll
      for (int i = 0; i < 8; ++i) {
        float mx = fmaxf(fmaxf(S[rt][0][i], S[rt][1][i]), fmaxf(S[rt][2][i], S[rt][3][i]));
#pragma unroll
        for (int m = 1; m < 16; m <<= 1) mx = fmaxf(mx, __shfl_xor(mx, m, 32));
        float mnew = fmaxf(m_i[rt][i], mx);
        alpha[rt][i] = __expf(m_i[rt][i] - mnew);
        m_i[rt][i] = mnew;
      }
    }

    // ---- P = exp(S - m) -> LDS (32x64 bf16 row-major) + row sums
#pragma unroll
    for (int rt = 0; rt < 2; ++rt) {
      float rs[8];
#pragma unroll
      for (int i = 0; i < 8; ++i) rs[i] = 0.f;
#pragma unroll
      for (int nt = 0; nt < 4; ++nt) {
#pragma unroll
        for (int i = 0; i < 8; ++i) {
          float p = __expf(S[rt][nt][i] - m_i[rt][i]);
          rs[i] += p;
          pshm[(rt * 16 + i + hg * 8) * 64 + nt * 16 + c] = f2bf(p);
        }
      }
#pragma unroll
      for (int i = 0; i < 8; ++i) {
#pragma unroll
        for (int m = 1; m < 16; m <<= 1) rs[i] += __shfl_xor(rs[i], m, 32);
        l_i[rt][i] = l_i[rt][i] * alpha[rt][i] + rs[i];
      }
#pragma unroll
      for (int dt = 0; dt < 4; ++dt)
#pragma unroll
        for (int i = 0; i < 8; ++i) O[rt][dt][i] *= alpha[rt][i];
    }

    __syncthreads();   // single-wave WG: orders LDS store->load

    // ---- O += P @ V : P A-frags from LDS, V B-frags pipelined from vT
    v16bf pa[2][2];
#pragma unroll
    for (int rt = 0; rt < 2; ++rt)
#pragma unroll
      for (int kk = 0; kk < 2; ++kk)
        pa[rt][kk] = load_afrag(&pshm[(rt * 16 + c) * 64 + kk * 32 + hg * 8]);

    v16bf vf0[2], vf1[2];
    vf0[0] = load_bfrag(vb + (size_t)c * MTOK + m0 + hg * 8);
    vf0[1] = load_bfrag(vb + (size_t)c * MTOK + m0 + 32 + hg * 8);
#pragma unroll
    for (int dt = 0; dt < 4; ++dt) {
      if (dt < 3) {
        const bf16_t* vp = vb + (size_t)((dt + 1) * 16 + c) * MTOK + m0 + hg * 8;
        vf1[0] = load_bfrag(vp);
        vf1[1] = load_bfrag(vp + 32);
      }
#pragma unroll
      for (int rt = 0; rt < 2; ++rt) {
        O[rt][dt] = wmma_bf16(pa[rt][0], vf0[0], O[rt][dt]);
        O[rt][dt] = wmma_bf16(pa[rt][1], vf0[1], O[rt][dt]);
      }
      vf0[0] = vf1[0]; vf0[1] = vf1[1];
    }
    __syncthreads();
  }

  // ---- normalize and store bf16 to attnout[b, n, h*64 + d]
  bf16_t* ob = attnout + ((size_t)(b * NTOK + qt * 32)) * DIM + h * HD;
#pragma unroll
  for (int rt = 0; rt < 2; ++rt) {
#pragma unroll
    for (int dt = 0; dt < 4; ++dt) {
#pragma unroll
      for (int i = 0; i < 8; ++i) {
        int row = rt * 16 + i + hg * 8;
        ob[(size_t)row * DIM + dt * 16 + c] = f2bf(O[rt][dt][i] / l_i[rt][i]);
      }
    }
  }
}

// ---------------------------------------------------------------- launch
extern "C" void kernel_launch(void* const* d_in, const int* in_sizes, int n_in,
                              void* d_out, int out_size, void* d_ws, size_t ws_size,
                              hipStream_t stream) {
  const float* x    = (const float*)d_in[0];
  const float* Wq   = (const float*)d_in[1];
  const float* Wkv  = (const float*)d_in[2];
  const float* sr_w = (const float*)d_in[3];
  const float* sr_b = (const float*)d_in[4];
  const float* ln_g = (const float*)d_in[5];
  const float* ln_b = (const float*)d_in[6];
  const float* Wp   = (const float*)d_in[7];
  const float* bp   = (const float*)d_in[8];
  float* out = (float*)d_out;

  char* ws = (char*)d_ws;
  size_t off = 0;
  auto alloc = [&](size_t bytes) {
    size_t o = off;
    off = (off + bytes + 255) & ~(size_t)255;
    return o;
  };
  const size_t NX = (size_t)NB * NTOK * DIM;          // 10,485,760
  const size_t NY = (size_t)NB * MTOK * DIM;          //  2,621,440
  bf16_t* xbf     = (bf16_t*)(ws + alloc(NX * 2));
  bf16_t* qbuf    = (bf16_t*)(ws + alloc(NX * 2));
  bf16_t* xpatch  = (bf16_t*)(ws + alloc((size_t)NB * MTOK * KCONV * 2));
  float*  ybuf    = (float*) (ws + alloc(NY * 4));
  bf16_t* xln     = (bf16_t*)(ws + alloc(NY * 2));
  bf16_t* kbuf    = (bf16_t*)(ws + alloc(NY * 2));
  bf16_t* vTbuf   = (bf16_t*)(ws + alloc(NY * 2));
  bf16_t* attnout = (bf16_t*)(ws + alloc(NX * 2));
  bf16_t* WqT     = (bf16_t*)(ws + alloc((size_t)DIM * DIM * 2));
  bf16_t* WkvT    = (bf16_t*)(ws + alloc((size_t)DIM * 2 * DIM * 2));
  bf16_t* WpT     = (bf16_t*)(ws + alloc((size_t)DIM * DIM * 2));
  bf16_t* srwT    = (bf16_t*)(ws + alloc((size_t)KCONV * DIM * 2));
  (void)ws_size; (void)in_sizes; (void)n_in; (void)out_size;

  // 1) precision conversions / layout prep
  cvt_f32_bf16<<<(int)((NX + 255) / 256), 256, 0, stream>>>(x, xbf, (int)NX);
  transpose_cvt<<<(DIM * DIM + 255) / 256, 256, 0, stream>>>(Wq, WqT, DIM, DIM);
  transpose_cvt<<<(DIM * 2 * DIM + 255) / 256, 256, 0, stream>>>(Wkv, WkvT, DIM, 2 * DIM);
  transpose_cvt<<<(DIM * DIM + 255) / 256, 256, 0, stream>>>(Wp, WpT, DIM, DIM);
  transpose_cvt<<<(KCONV * DIM + 255) / 256, 256, 0, stream>>>(sr_w, srwT, KCONV, DIM);
  patch_gather<<<(int)(((size_t)NB * MTOK * KCONV + 255) / 256), 256, 0, stream>>>(xbf, xpatch);

  // 2) q = x @ Wq                        (32768 x 320, K=320) -> bf16
  gemm_wmma<<<dim3(DIM / 64, NB * NTOK / 256), 256, 0, stream>>>(
      xbf, DIM, WqT, DIM, NB * NTOK, DIM, nullptr, nullptr, qbuf, nullptr, nullptr, 1);

  // 3) conv-as-GEMM: y = patches @ sr_w + sr_b   (8192 x 320, K=1280) -> f32
  gemm_wmma<<<dim3(DIM / 64, NB * MTOK / 256), 256, 0, stream>>>(
      xpatch, KCONV, srwT, KCONV, NB * MTOK, DIM, sr_b, ybuf, nullptr, nullptr, nullptr, 0);

  // 4) layernorm -> bf16
  layernorm_bf16<<<NB * MTOK / 4, 128, 0, stream>>>(ybuf, ln_g, ln_b, xln);

  // 5) kv = xln @ Wkv   (8192 x 640, K=320) -> k[b,h,m,d], vT[b,h,d,m]
  gemm_wmma<<<dim3(2 * DIM / 64, NB * MTOK / 256), 256, 0, stream>>>(
      xln, DIM, WkvT, DIM, NB * MTOK, 2 * DIM, nullptr, nullptr, nullptr, kbuf, vTbuf, 2);

  // 6) attention (flash, online softmax, 32 queries/wave) -> bf16 attnout
  attn_wmma<<<dim3(NTOK / 32, HEADS, NB), 32, 0, stream>>>(qbuf, kbuf, vTbuf, attnout);

  // 7) out = attnout @ Wp + bp  -> f32 d_out
  gemm_wmma<<<dim3(DIM / 64, NB * NTOK / 256), 256, 0, stream>>>(
      attnout, DIM, WpT, DIM, NB * NTOK, DIM, bp, out, nullptr, nullptr, nullptr, 0);
}
